// node_shuffle_30219389895260
// MI455X (gfx1250) — matrix operand order repaired and checked by reading, not compiled, not run on hardware
//
#include <hip/hip_runtime.h>

#define C      128
#define NPTS   4096
#define BATCH  8
#define ODIM   256
#define KNN    16
#define LPAD   144  // padded LDS row length (bf16 elems): 288B, 16B-divisible, bank-staggered

typedef __attribute__((ext_vector_type(16))) __bf16 v16bf;
typedef __attribute__((ext_vector_type(8)))  float  v8f;

__device__ __forceinline__ float  bf2f(__bf16 x) { return (float)x; }
__device__ __forceinline__ __bf16 f2bf(float x)  { return (__bf16)x; }

// ---- CDNA5 async global->LDS copy (VDST = LDS byte addr, 64-bit VADDR, SADDR=off) ----
__device__ __forceinline__ void async_copy_b128(const void* gsrc, void* ldst) {
  unsigned lds_addr = (unsigned)(unsigned long long)ldst;  // low 32 bits = LDS offset
  asm volatile("global_load_async_to_lds_b128 %0, %1, off"
               :: "v"(lds_addr), "v"(gsrc) : "memory");
}
__device__ __forceinline__ void wait_async0() {
  asm volatile("s_wait_asynccnt 0" ::: "memory");
}

// Stage a 128-col x 128-ch bf16 tile (rows contiguous, stride C in global) into LDS
// rows of stride LPAD. 16 b128 chunks per thread, 2048 total.
__device__ __forceinline__ void stage_tile_async(const __bf16* __restrict__ gbase,
                                                 __bf16* lbase, int tid) {
#pragma unroll
  for (int i = 0; i < 16; ++i) {
    const int t = i * 128 + tid;
    const int col = t >> 4;          // 0..127
    const int c0 = (t & 15) * 8;     // 0..120, 16B chunks
    async_copy_b128(gbase + (size_t)col * C + c0, lbase + col * LPAD + c0);
  }
}

// A-matrix fragment (16x32 bf16). Lane: m = lane&15, h = lane>>4.
// element e(0..7): k = 32kc+8h+e ; e(8..15): k = 32kc+16+8h+(e-8) (two 16B runs).
__device__ __forceinline__ v16bf load_a_frag(const __bf16* __restrict__ row, int kc, int h) {
  v16bf a;
  const __bf16* p0 = row + kc * 32 + 8 * h;
  const __bf16* p1 = row + kc * 32 + 16 + 8 * h;
#pragma unroll
  for (int e = 0; e < 8; ++e) { a[e] = p0[e]; a[8 + e] = p1[e]; }
  return a;
}

// B-matrix fragment (32x16 bf16). Lane: n = lane&15, h = lane>>4; element e(0..7): k = 32kc+8h+e ; e(8..15): k = 32kc+16+8h+(e-8).
__device__ __forceinline__ v16bf load_b_frag(const __bf16* __restrict__ col, int kc, int h) {
  v16bf b;
  const __bf16* p0 = col + kc * 32 + 8 * h;
  const __bf16* p1 = col + kc * 32 + 16 + 8 * h;
#pragma unroll
  for (int e = 0; e < 8; ++e) { b[e] = p0[e]; b[8 + e] = p1[e]; }
  return b;
}

// ---------------- Kernel 1: net = relu(w0 @ x + b0); emit netT (bf16) and sq ----------------
__global__ __launch_bounds__(128) void net_kernel(const float* __restrict__ x,
                                                  const float* __restrict__ w0,
                                                  const float* __restrict__ b0,
                                                  __bf16* __restrict__ netT,
                                                  float* __restrict__ sq) {
  __shared__ __align__(16) __bf16 w0s[C][136];   // row-major, padded
  __shared__ __align__(16) __bf16 xT[128][136];  // xT[n_local][c]
  __shared__ float sqs[128];
  const int tid = threadIdx.x;
  const int b = blockIdx.y;
  const int n0 = blockIdx.x * 128;

  for (int r = 0; r < C; ++r) w0s[r][tid] = f2bf(w0[r * C + tid]);
  for (int c = 0; c < C; ++c) xT[tid][c] = f2bf(x[((size_t)b * C + c) * NPTS + n0 + tid]);
  sqs[tid] = 0.f;
  __syncthreads();

  const int wave = tid >> 5, lane = tid & 31, ln = lane & 15, h = lane >> 4;
#pragma unroll
  for (int mi = 0; mi < 2; ++mi) {
    const int mt = wave * 2 + mi;  // 8 M-tiles over 4 waves
    v16bf a[4];
#pragma unroll
    for (int kc = 0; kc < 4; ++kc) a[kc] = load_a_frag(&w0s[mt * 16 + ln][0], kc, h);
#pragma unroll
    for (int nst = 0; nst < 8; ++nst) {
      v8f acc = {};
#pragma unroll
      for (int kc = 0; kc < 4; ++kc) {
        v16bf bb = load_b_frag(&xT[nst * 16 + ln][0], kc, h);
        acc = __builtin_amdgcn_wmma_f32_16x16x32_bf16(false, a[kc], false, bb, (short)0, acc,
                                                      false, false);
      }
      const int col_l = nst * 16 + ln;
      const int col = n0 + col_l;
      float ss = 0.f;
#pragma unroll
      for (int v = 0; v < 8; ++v) {
        const int row = mt * 16 + v + 8 * h;  // D layout: m = v + 8h
        float val = acc[v] + b0[row];
        val = val > 0.f ? val : 0.f;
        ss += val * val;
        netT[((size_t)b * NPTS + col) * C + row] = f2bf(val);
      }
      atomicAdd(&sqs[col_l], ss);  // ds_add_f32
    }
  }
  __syncthreads();
  sq[(size_t)b * NPTS + n0 + tid] = sqs[tid];
}

// ---------------- Kernel 2: ctrb = w2 @ net + b2 + b1 (async-staged B tile) ----------------
__global__ __launch_bounds__(128) void ctr_kernel(const __bf16* __restrict__ netT,
                                                  const float* __restrict__ w2,
                                                  const float* __restrict__ b1,
                                                  const float* __restrict__ b2,
                                                  float* __restrict__ ctrb) {
  __shared__ __align__(16) __bf16 w2s[ODIM][136];
  __shared__ __align__(16) __bf16 colS[128][LPAD];
  const int tid = threadIdx.x, b = blockIdx.y, n0 = blockIdx.x * 128;
  // kick off async stage of the column tile, convert w2 while it flies
  stage_tile_async(&netT[((size_t)b * NPTS + n0) * C], &colS[0][0], tid);
  for (int r = 0; r < ODIM; r += 2) {  // 256 rows, 128 threads
    w2s[r][tid] = f2bf(w2[r * C + tid]);
    w2s[r + 1][tid] = f2bf(w2[(r + 1) * C + tid]);
  }
  wait_async0();
  __syncthreads();
  const int wave = tid >> 5, lane = tid & 31, ln = lane & 15, h = lane >> 4;
#pragma unroll
  for (int mi = 0; mi < 4; ++mi) {
    const int mt = wave * 4 + mi;  // 16 M-tiles over 4 waves
    v16bf a[4];
#pragma unroll
    for (int kc = 0; kc < 4; ++kc) a[kc] = load_a_frag(&w2s[mt * 16 + ln][0], kc, h);
#pragma unroll
    for (int nst = 0; nst < 8; ++nst) {
      const int coll = nst * 16 + ln;
      v8f acc = {};
#pragma unroll
      for (int kc = 0; kc < 4; ++kc) {
        v16bf bb = load_b_frag(&colS[coll][0], kc, h);
        acc = __builtin_amdgcn_wmma_f32_16x16x32_bf16(false, a[kc], false, bb, (short)0, acc,
                                                      false, false);
      }
#pragma unroll
      for (int v = 0; v < 8; ++v) {
        const int row = mt * 16 + v + 8 * h;
        ctrb[((size_t)b * ODIM + row) * NPTS + n0 + coll] = acc[v] + b1[row] + b2[row];
      }
    }
  }
}

// ------ Kernel 3: fused Gram + kNN top-16, double-buffered async B-tile staging ------
__global__ __launch_bounds__(128) void knn_kernel(const __bf16* __restrict__ netT,
                                                  const float* __restrict__ sq,
                                                  int* __restrict__ idxb) {
  __shared__ __align__(16) __bf16 dbS[2][128][LPAD];  // ping-pong database tiles
  __shared__ float distS[128][132];
  __shared__ float sqdS[128];
  const int tid = threadIdx.x, b = blockIdx.y, q0 = blockIdx.x * 128;
  const int wave = tid >> 5, lane = tid & 31, ln = lane & 15, h = lane >> 4;
  const __bf16* nbase = &netT[(size_t)b * NPTS * C];

  // Persistent A fragments: the workgroup's 128 query rows.
  v16bf a[2][4];
#pragma unroll
  for (int mi = 0; mi < 2; ++mi) {
    const int mt = wave * 2 + mi;
    const __bf16* rp = nbase + (size_t)(q0 + mt * 16 + ln) * C;
#pragma unroll
    for (int kc = 0; kc < 4; ++kc) a[mi][kc] = load_a_frag(rp, kc, h);
  }
  const float sqq = sq[(size_t)b * NPTS + q0 + tid];

  float bestd[KNN];
  int   besti[KNN];
#pragma unroll
  for (int s = 0; s < KNN; ++s) { bestd[s] = 3.0e38f; besti[s] = 0; }
  float wd = 3.0e38f;
  int   wi = 0;

  // prologue: stage tile 0
  stage_tile_async(nbase, &dbS[0][0][0], tid);

  for (int dt = 0; dt < NPTS / 128; ++dt) {
    const int d0 = dt * 128;
    const int pb = dt & 1;
    sqdS[tid] = sq[(size_t)b * NPTS + d0 + tid];
    wait_async0();        // tile dt resident (per-wave), barrier makes it global
    __syncthreads();
    if (dt + 1 < NPTS / 128)  // overlap next tile's copy with this tile's WMMA work
      stage_tile_async(nbase + (size_t)(d0 + 128) * C, &dbS[1 - pb][0][0], tid);

#pragma unroll
    for (int mi = 0; mi < 2; ++mi) {
      const int mt = wave * 2 + mi;
#pragma unroll
      for (int nst = 0; nst < 8; ++nst) {
        const int coll = nst * 16 + ln;
        v8f acc = {};
#pragma unroll
        for (int kc = 0; kc < 4; ++kc) {
          v16bf bb = load_b_frag(&dbS[pb][coll][0], kc, h);
          acc = __builtin_amdgcn_wmma_f32_16x16x32_bf16(false, a[mi][kc], false, bb, (short)0,
                                                        acc, false, false);
        }
        const float sj = sqdS[coll];
#pragma unroll
        for (int v = 0; v < 8; ++v) distS[mt * 16 + v + 8 * h][coll] = sj - 2.0f * acc[v];
      }
    }
    __syncthreads();
    const int i = q0 + tid;
    for (int j = 0; j < 128; ++j) {
      const int jg = d0 + j;
      const float d = distS[tid][j] + sqq;
      if (jg != i && d < wd) {
        bestd[wi] = d;
        besti[wi] = jg;
        wd = -1.f;
#pragma unroll
        for (int s = 0; s < KNN; ++s)
          if (bestd[s] > wd) { wd = bestd[s]; wi = s; }
      }
    }
    // next iteration's top barrier separates this scan from distS/sqdS overwrite
  }
#pragma unroll
  for (int s = 0; s < KNN; ++s) idxb[((size_t)b * NPTS + q0 + tid) * KNN + s] = besti[s];
}

// ---------------- Kernel 4: edge GEMM + relu + max over k ----------------
__global__ __launch_bounds__(128) void edge_kernel(const __bf16* __restrict__ netT,
                                                   const float* __restrict__ w1,
                                                   const int* __restrict__ idxb,
                                                   const float* __restrict__ ctrb,
                                                   float* __restrict__ out) {
  __shared__ __align__(16) __bf16 w1s[ODIM][136];
  __shared__ __align__(16) __bf16 diffT[4][KNN][136];  // per-wave neighbor-diff tile, [j][c]
  const int tid = threadIdx.x;
  const int wave = tid >> 5, lane = tid & 31, ln = lane & 15, h = lane >> 4;
  for (int r = 0; r < ODIM; r += 2) {
    w1s[r][tid] = f2bf(w1[r * C + tid]);
    w1s[r + 1][tid] = f2bf(w1[(r + 1) * C + tid]);
  }
  const int p = blockIdx.x * 4 + wave;  // one point per wave
  const int b = p / NPTS, n = p % NPTS;
  {
    const int nbr = idxb[(size_t)p * KNN + ln];
    const __bf16* cc = &netT[((size_t)b * NPTS + n) * C];
    const __bf16* nc = &netT[((size_t)b * NPTS + nbr) * C];
    for (int i2 = 0; i2 < 64; ++i2) {  // lane pair (h) splits the 128 channels
      const int c = h * 64 + i2;
      diffT[wave][ln][c] = f2bf(bf2f(nc[c]) - bf2f(cc[c]));
    }
  }
  __syncthreads();

  v16bf bb[4];
#pragma unroll
  for (int kc = 0; kc < 4; ++kc) bb[kc] = load_b_frag(&diffT[wave][ln][0], kc, h);

  for (int ot = 0; ot < 16; ++ot) {  // 256 output channels
    v16bf a[4];
#pragma unroll
    for (int kc = 0; kc < 4; ++kc) a[kc] = load_a_frag(&w1s[ot * 16 + ln][0], kc, h);
    v8f acc = {};
#pragma unroll
    for (int kc = 0; kc < 4; ++kc)
      acc = __builtin_amdgcn_wmma_f32_16x16x32_bf16(false, a[kc], false, bb[kc], (short)0, acc,
                                                    false, false);
#pragma unroll
    for (int v = 0; v < 8; ++v) {
      const int o = ot * 16 + v + 8 * h;
      float e = acc[v] + ctrb[((size_t)b * ODIM + o) * NPTS + n];
      e = e > 0.f ? e : 0.f;
      // max over the 16 neighbor columns (lanes within each 16-lane group)
      e = fmaxf(e, __shfl_xor(e, 1, 32));
      e = fmaxf(e, __shfl_xor(e, 2, 32));
      e = fmaxf(e, __shfl_xor(e, 4, 32));
      e = fmaxf(e, __shfl_xor(e, 8, 32));
      if (ln == 0) out[((size_t)b * ODIM + o) * NPTS + n] = e;  // reshape is a no-op
    }
  }
}

extern "C" void kernel_launch(void* const* d_in, const int* in_sizes, int n_in,
                              void* d_out, int out_size, void* d_ws, size_t ws_size,
                              hipStream_t stream) {
  (void)in_sizes; (void)n_in; (void)out_size; (void)ws_size;
  const float* x  = (const float*)d_in[0];
  const float* w0 = (const float*)d_in[1];
  const float* b0 = (const float*)d_in[2];
  const float* w1 = (const float*)d_in[3];
  const float* b1 = (const float*)d_in[4];
  const float* w2 = (const float*)d_in[5];
  const float* b2 = (const float*)d_in[6];
  float* out = (float*)d_out;

  char* ws = (char*)d_ws;
  size_t off = 0;
  __bf16* netT = (__bf16*)(ws + off); off += (size_t)BATCH * NPTS * C * sizeof(__bf16);
  float*  sq   = (float*)(ws + off);  off += (size_t)BATCH * NPTS * sizeof(float);
  float*  ctrb = (float*)(ws + off);  off += (size_t)BATCH * ODIM * NPTS * sizeof(float);
  int*    idxb = (int*)(ws + off);    off += (size_t)BATCH * NPTS * KNN * sizeof(int);

  dim3 blk(128);
  net_kernel<<<dim3(NPTS / 128, BATCH), blk, 0, stream>>>(x, w0, b0, netT, sq);
  ctr_kernel<<<dim3(NPTS / 128, BATCH), blk, 0, stream>>>(netT, w2, b1, b2, ctrb);
  knn_kernel<<<dim3(NPTS / 128, BATCH), blk, 0, stream>>>(netT, sq, idxb);
  edge_kernel<<<(BATCH * NPTS) / 4, blk, 0, stream>>>(netT, w1, idxb, ctrb, out);
}
